// QwenSparseMoeBlock_10685878632759
// MI455X (gfx1250) — matrix operand order repaired and checked by reading, not compiled, not run on hardware
//
#include <hip/hip_runtime.h>
#include <hip/hip_bf16.h>
#include <cstdint>

// ---------------- problem dims (compile-time) ----------------
#define T_TOK   2048
#define H_DIM   2048
#define N_EXP   8
#define I_DIM   1408
#define ISH_DIM 5632

typedef __attribute__((ext_vector_type(16))) __bf16 v16bf;
typedef __attribute__((ext_vector_type(8)))  float  v8f;
using bf16 = __bf16;

union Frag16 { v16bf v; uint4 q[2]; };

static __device__ __forceinline__ uint32_t pack_bf16x2(float a, float b) {
    union { bf16 h[2]; uint32_t u; } p;
    p.h[0] = (bf16)a; p.h[1] = (bf16)b;
    return p.u;
}

// fast sigmoid: v_exp_f32 + v_rcp_f32 (no IEEE divide sequence)
static __device__ __forceinline__ float fast_sigmoid(float x) {
    return __builtin_amdgcn_rcpf(1.f + __expf(-x));
}

// CDNA5 async global->LDS DMA (per-lane 16B copy, tracked by ASYNCcnt).
// VDST = 32-bit LDS byte address, VADDR = 64-bit global address (GV mode).
static __device__ __forceinline__ void async_copy_b128(uint32_t lds_off, const bf16* gptr) {
    asm volatile("global_load_async_to_lds_b128 %0, %1, off"
                 :: "v"(lds_off), "v"((uint64_t)(uintptr_t)gptr)
                 : "memory");
}
static __device__ __forceinline__ void wait_asynccnt0() {
    asm volatile("s_wait_asynccnt 0x0" ::: "memory");
}
static __device__ __forceinline__ uint32_t lds_addr(const void* p) {
    return (uint32_t)(uintptr_t)p;   // addrspace(3) offset lives in low 32 bits
}

// ---------------------------------------------------------------------------
// Router: logits, softmax, deterministic top-2, dense routing weights,
// shared-expert sigmoid gate. One wave per token (wave32).
// ---------------------------------------------------------------------------
__global__ __launch_bounds__(256)
void router_kernel(const float* __restrict__ x,
                   const float* __restrict__ rw,    // [H, E]
                   const float* __restrict__ segw,  // [H, 1]
                   float* __restrict__ logits_out,  // d_out + T*H
                   float* __restrict__ routing,     // [T, E]
                   float* __restrict__ sgate,       // [T]
                   int*   __restrict__ topk_ids)    // [T, 2]
{
    const int wid  = threadIdx.x >> 5;
    const int lane = threadIdx.x & 31;
    const int t = blockIdx.x * 8 + wid;
    if (t >= T_TOK) return;

    float acc[N_EXP];
    float accs = 0.f;
#pragma unroll
    for (int e = 0; e < N_EXP; ++e) acc[e] = 0.f;

    for (int k = lane; k < H_DIM; k += 32) {
        const float xv = x[(size_t)t * H_DIM + k];
        const float* w = rw + (size_t)k * N_EXP;
#pragma unroll
        for (int e = 0; e < N_EXP; ++e) acc[e] += xv * w[e];
        accs += xv * segw[k];
    }
#pragma unroll
    for (int e = 0; e < N_EXP; ++e)
        for (int o = 16; o > 0; o >>= 1) acc[e] += __shfl_xor(acc[e], o, 32);
    for (int o = 16; o > 0; o >>= 1) accs += __shfl_xor(accs, o, 32);

    if (lane == 0) {
        float m = acc[0];
#pragma unroll
        for (int e = 1; e < N_EXP; ++e) m = fmaxf(m, acc[e]);
        float p[N_EXP], s = 0.f;
#pragma unroll
        for (int e = 0; e < N_EXP; ++e) { p[e] = __expf(acc[e] - m); s += p[e]; }
        const float inv = 1.f / s;   // exact: feeds routing weights
        int i0 = 0;
#pragma unroll
        for (int e = 1; e < N_EXP; ++e) if (p[e] > p[i0]) i0 = e;
        int i1 = (i0 == 0) ? 1 : 0;
#pragma unroll
        for (int e = 0; e < N_EXP; ++e) if (e != i0 && p[e] > p[i1]) i1 = e;
#pragma unroll
        for (int e = 0; e < N_EXP; ++e) {
            logits_out[(size_t)t * N_EXP + e] = acc[e];
            routing[(size_t)t * N_EXP + e] = (e == i0 || e == i1) ? p[e] * inv : 0.f;
        }
        topk_ids[t * 2 + 0] = i0;
        topk_ids[t * 2 + 1] = i1;
        sgate[t] = fast_sigmoid(accs);
    }
}

// Deterministic per-expert token lists (order-stable across graph replays).
__global__ void build_lists_kernel(const int* __restrict__ topk_ids,
                                   int* __restrict__ lists,   // [E, T]
                                   int* __restrict__ counts)  // [E]
{
    const int e = threadIdx.x;
    if (e >= N_EXP) return;
    int c = 0;
    for (int t = 0; t < T_TOK; ++t)
        if (topk_ids[t * 2] == e || topk_ids[t * 2 + 1] == e)
            lists[e * T_TOK + c++] = t;
    counts[e] = c;
}

// f32 -> bf16 activation convert (vectorized)
__global__ __launch_bounds__(256)
void cvt_bf16_kernel(const float* __restrict__ in, bf16* __restrict__ out, int n4)
{
    const int i = blockIdx.x * blockDim.x + threadIdx.x;
    if (i >= n4) return;
    const float4 v = ((const float4*)in)[i];
    union { bf16 b[4]; ushort4 u; } pk;
    pk.b[0] = (bf16)v.x; pk.b[1] = (bf16)v.y; pk.b[2] = (bf16)v.z; pk.b[3] = (bf16)v.w;
    ((ushort4*)out)[i] = pk.u;
}

// ---------------------------------------------------------------------------
// Dual-B GEMM: C_gate = A@Bg, C_up = A@Bu over shared A tile, fused
// hidden = up * silu(gate) [* routing] epilogue, bf16 output.
// Block: 256 thr = 8 waves, tile M=128 x N=64 (per set), K-step 32.
// Double-buffered LDS; A tile staged via async global->LDS DMA.
// GATHER: A rows gathered via per-expert token list, routing weight folded in.
// ---------------------------------------------------------------------------
template<bool GATHER>
__global__ __launch_bounds__(256)
void gemm_gateup_kernel(const bf16*  __restrict__ A,      // [T, K] bf16
                        const float* __restrict__ Bg,     // [K, ldb] f32
                        const float* __restrict__ Bu,     // [K, ldb] f32
                        int ldb, int Kdim,
                        const int*   __restrict__ lists,  // [E, T]
                        const int*   __restrict__ counts, // [E]
                        const float* __restrict__ routing,// [T, E]
                        bf16* __restrict__ Hout, int ldh)
{
    __shared__ __align__(16) bf16 sA [2][128 * 32];
    __shared__ __align__(16) bf16 sBg[2][ 64 * 32];
    __shared__ __align__(16) bf16 sBu[2][ 64 * 32];

    const int e = GATHER ? blockIdx.z : 0;
    const int* list = GATHER ? (lists + e * T_TOK) : nullptr;
    const int rows = GATHER ? counts[e] : T_TOK;

    const int m0 = blockIdx.x * 128;
    if (m0 >= rows) return;
    const int n0 = blockIdx.y * 64;

    const float* bg = Bg + (GATHER ? (size_t)e * Kdim * ldb : 0);
    const float* bu = Bu + (GATHER ? (size_t)e * Kdim * ldb : 0);

    const int tid  = threadIdx.x;
    const int lane = tid & 31;
    const int wid  = tid >> 5;
    const int wm   = wid >> 1;   // 0..3 -> 32-row slice
    const int wn   = wid & 1;    // 0..1 -> 32-col slice
    const int lhalf = lane >> 4;
    const int lrow  = lane & 15;

    // --- A staging plan: 128x32 bf16 = 512 x 16B; 2 async copies per thread ---
    const bf16* asrc[2];
    int adst[2];
#pragma unroll
    for (int i = 0; i < 2; ++i) {
        const int idx = tid + i * 256;
        const int r = idx >> 2, q = idx & 3;
        const int row = m0 + r;
        const int tok = GATHER ? list[min(row, rows - 1)] : min(row, rows - 1);
        asrc[i] = A + (size_t)tok * Kdim + q * 8;
        adst[i] = r * 32 + q * 8;
    }
    // --- B staging plan: 32(k) x 64(n); thread owns column n, 8 k values ---
    const int bn = tid & 63;
    const int bk = (tid >> 6) * 8;

    auto stageA = [&](int k0, int buf) {
#pragma unroll
        for (int i = 0; i < 2; ++i)
            async_copy_b128(lds_addr(&sA[buf][adst[i]]), asrc[i] + k0);
    };
    auto stageB = [&](int k0, int buf) {
        const float* pg = bg + (size_t)(k0 + bk) * ldb + (n0 + bn);
        const float* pu = bu + (size_t)(k0 + bk) * ldb + (n0 + bn);
        if (k0 + 32 < Kdim) {
            __builtin_prefetch(pg + (size_t)32 * ldb, 0, 1);
            __builtin_prefetch(pu + (size_t)32 * ldb, 0, 1);
        }
#pragma unroll
        for (int j = 0; j < 8; j += 2) {
            const float g0 = pg[(size_t)j * ldb], g1 = pg[(size_t)(j + 1) * ldb];
            const float u0 = pu[(size_t)j * ldb], u1 = pu[(size_t)(j + 1) * ldb];
            *(uint32_t*)&sBg[buf][bn * 32 + bk + j] = pack_bf16x2(g0, g1);
            *(uint32_t*)&sBu[buf][bn * 32 + bk + j] = pack_bf16x2(u0, u1);
        }
    };

    v8f accG[2][2] = {};
    v8f accU[2][2] = {};

    const int nk = Kdim / 32;
    stageA(0, 0);
    stageB(0, 0);
    wait_asynccnt0();
    __syncthreads();

    for (int kt = 0; kt < nk; ++kt) {
        const int cur = kt & 1, nxt = cur ^ 1;
        const bool more = (kt + 1) < nk;
        if (more) { stageA((kt + 1) * 32, nxt); stageB((kt + 1) * 32, nxt); }

        // fragment loads per CDNA5 WMMA VGPR layouts
        Frag16 af[2], bgf[2], buf_[2];
#pragma unroll
        for (int mt = 0; mt < 2; ++mt) {
            const bf16* p = &sA[cur][(wm * 32 + mt * 16 + lrow) * 32 + lhalf * 8];
            af[mt].q[0] = *(const uint4*)p;
            af[mt].q[1] = *(const uint4*)(p + 16);
        }
#pragma unroll
        for (int nt = 0; nt < 2; ++nt) {
            const int n = (wn * 32 + nt * 16 + lrow) * 32 + lhalf * 8;
            bgf[nt].q[0] = *(const uint4*)&sBg[cur][n];
            bgf[nt].q[1] = *(const uint4*)&sBg[cur][n + 16];
            buf_[nt].q[0] = *(const uint4*)&sBu[cur][n];
            buf_[nt].q[1] = *(const uint4*)&sBu[cur][n + 16];
        }
#pragma unroll
        for (int mt = 0; mt < 2; ++mt)
#pragma unroll
            for (int nt = 0; nt < 2; ++nt) {
                accG[mt][nt] = __builtin_amdgcn_wmma_f32_16x16x32_bf16(
                    false, af[mt].v, false, bgf[nt].v, (short)0, accG[mt][nt], false, false);
                accU[mt][nt] = __builtin_amdgcn_wmma_f32_16x16x32_bf16(
                    false, af[mt].v, false, buf_[nt].v, (short)0, accU[mt][nt], false, false);
            }

        if (more) wait_asynccnt0();
        __syncthreads();
    }

    // epilogue: hidden = up * silu(gate) [* routing], bf16 store
    bf16* hbase = Hout + (GATHER ? (size_t)e * T_TOK * ldh : 0);
#pragma unroll
    for (int mt = 0; mt < 2; ++mt) {
#pragma unroll
        for (int i = 0; i < 8; ++i) {
            const int gm = m0 + wm * 32 + mt * 16 + lhalf * 8 + i;
            if (gm >= rows) continue;
            float scale = 1.f;
            if (GATHER) scale = routing[(size_t)list[gm] * N_EXP + e];
#pragma unroll
            for (int nt = 0; nt < 2; ++nt) {
                const float g = accG[mt][nt][i];
                const float u = accU[mt][nt][i];
                const float h = scale * u * g * fast_sigmoid(g);   // silu via v_rcp_f32
                hbase[(size_t)gm * ldh + n0 + wn * 32 + nt * 16 + lrow] = (bf16)h;
            }
        }
    }
}

// ---------------------------------------------------------------------------
// Down-projection GEMM: [rows,K] bf16 x [K,H] f32(->bf16).
// Double-buffered LDS; async A staging.
// EXPERT: scatter atomicAdd into out at gathered token rows.
// !EXPERT (shared): out[t,h] = sgate[t] * val (plain store, runs first).
// ---------------------------------------------------------------------------
template<bool EXPERT>
__global__ __launch_bounds__(256)
void gemm_out_kernel(const bf16*  __restrict__ A,      // hidden or sh_mid
                     const float* __restrict__ B,      // [K, H]
                     int Kdim,
                     const int*   __restrict__ lists,
                     const int*   __restrict__ counts,
                     const float* __restrict__ sgate,
                     float* __restrict__ out)          // [T, H]
{
    __shared__ __align__(16) bf16 sA[2][128 * 32];
    __shared__ __align__(16) bf16 sB[2][ 64 * 32];

    const int e = EXPERT ? blockIdx.z : 0;
    const int* list = EXPERT ? (lists + e * T_TOK) : nullptr;
    const int rows = EXPERT ? counts[e] : T_TOK;

    const int m0 = blockIdx.x * 128;
    if (m0 >= rows) return;
    const int n0 = blockIdx.y * 64;

    const bf16*  a = A + (EXPERT ? (size_t)e * T_TOK * Kdim : 0);
    const float* b = B + (EXPERT ? (size_t)e * Kdim * H_DIM : 0);

    const int tid  = threadIdx.x;
    const int lane = tid & 31;
    const int wid  = tid >> 5;
    const int wm   = wid >> 1;
    const int wn   = wid & 1;
    const int lhalf = lane >> 4;
    const int lrow  = lane & 15;

    const bf16* asrc[2];
    int adst[2];
#pragma unroll
    for (int i = 0; i < 2; ++i) {
        const int idx = tid + i * 256;
        const int r = idx >> 2, q = idx & 3;
        const int row = min(m0 + r, rows - 1);
        asrc[i] = a + (size_t)row * Kdim + q * 8;
        adst[i] = r * 32 + q * 8;
    }
    const int bn = tid & 63;
    const int bk = (tid >> 6) * 8;

    auto stageA = [&](int k0, int buf) {
#pragma unroll
        for (int i = 0; i < 2; ++i)
            async_copy_b128(lds_addr(&sA[buf][adst[i]]), asrc[i] + k0);
    };
    auto stageB = [&](int k0, int buf) {
        const float* pb = b + (size_t)(k0 + bk) * H_DIM + (n0 + bn);
        if (k0 + 32 < Kdim) __builtin_prefetch(pb + (size_t)32 * H_DIM, 0, 1);
#pragma unroll
        for (int j = 0; j < 8; j += 2) {
            const float b0 = pb[(size_t)j * H_DIM], b1 = pb[(size_t)(j + 1) * H_DIM];
            *(uint32_t*)&sB[buf][bn * 32 + bk + j] = pack_bf16x2(b0, b1);
        }
    };

    v8f acc[2][2] = {};

    const int nk = Kdim / 32;
    stageA(0, 0);
    stageB(0, 0);
    wait_asynccnt0();
    __syncthreads();

    for (int kt = 0; kt < nk; ++kt) {
        const int cur = kt & 1, nxt = cur ^ 1;
        const bool more = (kt + 1) < nk;
        if (more) { stageA((kt + 1) * 32, nxt); stageB((kt + 1) * 32, nxt); }

        Frag16 af[2], bf[2];
#pragma unroll
        for (int mt = 0; mt < 2; ++mt) {
            const bf16* p = &sA[cur][(wm * 32 + mt * 16 + lrow) * 32 + lhalf * 8];
            af[mt].q[0] = *(const uint4*)p;
            af[mt].q[1] = *(const uint4*)(p + 16);
        }
#pragma unroll
        for (int nt = 0; nt < 2; ++nt) {
            const int n = (wn * 32 + nt * 16 + lrow) * 32 + lhalf * 8;
            bf[nt].q[0] = *(const uint4*)&sB[cur][n];
            bf[nt].q[1] = *(const uint4*)&sB[cur][n + 16];
        }
#pragma unroll
        for (int mt = 0; mt < 2; ++mt)
#pragma unroll
            for (int nt = 0; nt < 2; ++nt)
                acc[mt][nt] = __builtin_amdgcn_wmma_f32_16x16x32_bf16(
                    false, af[mt].v, false, bf[nt].v, (short)0, acc[mt][nt], false, false);

        if (more) wait_asynccnt0();
        __syncthreads();
    }

#pragma unroll
    for (int mt = 0; mt < 2; ++mt) {
#pragma unroll
        for (int i = 0; i < 8; ++i) {
            const int gm = m0 + wm * 32 + mt * 16 + lhalf * 8 + i;
            if (gm >= rows) continue;
            const int tok = EXPERT ? list[gm] : gm;
            const float sg = EXPERT ? 1.f : sgate[gm];
#pragma unroll
            for (int nt = 0; nt < 2; ++nt) {
                const int col = n0 + wn * 32 + nt * 16 + lrow;
                const float v = acc[mt][nt][i];
                if (EXPERT) atomicAdd(&out[(size_t)tok * H_DIM + col], v);
                else        out[(size_t)tok * H_DIM + col] = v * sg;
            }
        }
    }
}

// ---------------------------------------------------------------------------
extern "C" void kernel_launch(void* const* d_in, const int* in_sizes, int n_in,
                              void* d_out, int out_size, void* d_ws, size_t ws_size,
                              hipStream_t stream)
{
    (void)in_sizes; (void)n_in; (void)out_size; (void)ws_size;

    const float* x    = (const float*)d_in[0]; // [T, H]
    const float* rw   = (const float*)d_in[1]; // [H, E]
    const float* egu  = (const float*)d_in[2]; // [E, H, 2I]
    const float* eow  = (const float*)d_in[3]; // [E, I, H]
    const float* sgw  = (const float*)d_in[4]; // [H, ISH]
    const float* suw  = (const float*)d_in[5]; // [H, ISH]
    const float* sow  = (const float*)d_in[6]; // [ISH, H]
    const float* segw = (const float*)d_in[7]; // [H, 1]

    float* out    = (float*)d_out;
    float* logits = out + (size_t)T_TOK * H_DIM;

    char* ws = (char*)d_ws;
    auto alloc = [&](size_t bytes) -> char* {
        char* p = ws; ws += (bytes + 255) & ~(size_t)255; return p;
    };
    bf16*  xb      = (bf16*) alloc((size_t)T_TOK * H_DIM * 2);
    bf16*  shmid   = (bf16*) alloc((size_t)T_TOK * ISH_DIM * 2);
    bf16*  hidden  = (bf16*) alloc((size_t)N_EXP * T_TOK * I_DIM * 2);
    float* routing = (float*)alloc((size_t)T_TOK * N_EXP * 4);
    float* sgate   = (float*)alloc((size_t)T_TOK * 4);
    int*   topk    = (int*)  alloc((size_t)T_TOK * 2 * 4);
    int*   lists   = (int*)  alloc((size_t)N_EXP * T_TOK * 4);
    int*   counts  = (int*)  alloc(256);

    // 1) router + shared sigmoid gate + logits to d_out tail
    router_kernel<<<T_TOK / 8, 256, 0, stream>>>(x, rw, segw, logits, routing, sgate, topk);
    // 2) deterministic expert token lists
    build_lists_kernel<<<1, 32, 0, stream>>>(topk, lists, counts);
    // 3) activations to bf16
    cvt_bf16_kernel<<<(T_TOK * H_DIM / 4 + 255) / 256, 256, 0, stream>>>(x, xb, T_TOK * H_DIM / 4);
    // 4) shared expert gate/up + SwiGLU  [T,H] x [H,ISH] (dual B)
    gemm_gateup_kernel<false><<<dim3(T_TOK / 128, ISH_DIM / 64), 256, 0, stream>>>(
        xb, sgw, suw, ISH_DIM, H_DIM, nullptr, nullptr, nullptr, shmid, ISH_DIM);
    // 5) routed experts gate/up + SwiGLU, gathered rows, routing folded in
    gemm_gateup_kernel<true><<<dim3(T_TOK / 128, I_DIM / 64, N_EXP), 256, 0, stream>>>(
        xb, egu, egu + I_DIM, 2 * I_DIM, H_DIM, lists, counts, routing, hidden, I_DIM);
    // 6) shared down-proj, writes out = sgate * (mid @ W)
    gemm_out_kernel<false><<<dim3(T_TOK / 128, H_DIM / 64), 256, 0, stream>>>(
        shmid, sow, ISH_DIM, nullptr, nullptr, sgate, out);
    // 7) expert down-proj, scatter-accumulate into out
    gemm_out_kernel<true><<<dim3(T_TOK / 128, H_DIM / 64, N_EXP), 256, 0, stream>>>(
        hidden, eow, I_DIM, lists, counts, nullptr, out);
}
